// MHA_35527969472530
// MI455X (gfx1250) — matrix operand order repaired and checked by reading, not compiled, not run on hardware
//
#include <hip/hip_runtime.h>
#include <hip/hip_bf16.h>
#include <math.h>

typedef __bf16 bf16;
typedef __attribute__((ext_vector_type(8)))  bf16  bf16x8;
typedef __attribute__((ext_vector_type(16))) bf16  bf16x16;
typedef __attribute__((ext_vector_type(8)))  float floatx8;
typedef __attribute__((ext_vector_type(4)))  unsigned int uint32x4;
typedef __attribute__((ext_vector_type(8)))  unsigned int uint32x8;

#define S_LEN 2048
#define BATCH 2
#define FDIM 1024
#define HEADS 16
#define DHEAD 64
#define NH (BATCH*HEADS)        // 32
#define ROWS (S_LEN*BATCH)      // 4096
#define P3 (3*FDIM)             // 3072

// ---------------- fragment helpers (CDNA5 WMMA bf16 16x16x32 layouts) -------

// A fragment 16x32 (M x K), source row-major, p -> (row0, col0), ld in elems.
// lane<16: M=lane, elems = K{0..7, 16..23}; lane>=16: K{8..15, 24..31}.
__device__ __forceinline__ bf16x16 frag_a(const bf16* p, int ld) {
    int lane = threadIdx.x & 31;
    int r = lane & 15, h = lane >> 4;
    const bf16* base = p + (size_t)r * ld + h * 8;
    bf16x8 lo = *(const bf16x8*)(base);
    bf16x8 hi = *(const bf16x8*)(base + 16);
    bf16x16 out;
#pragma unroll
    for (int i = 0; i < 8; ++i) { out[i] = lo[i]; out[i + 8] = hi[i]; }
    return out;
}

// B fragment 32x16 (K x N), source is B^T row-major (rows = N), pT -> (n0, k0).
// lane<16: N=lane, elems = K{0..7, 16..23}; lane>=16: K{8..15, 24..31}.
__device__ __forceinline__ bf16x16 frag_b(const bf16* pT, int ld) {
    int lane = threadIdx.x & 31;
    int r = lane & 15, h = lane >> 4;
    const bf16* base = pT + (size_t)r * ld + h * 8;
    bf16x8 lo = *(const bf16x8*)(base);
    bf16x8 hi = *(const bf16x8*)(base + 16);
    bf16x16 out;
#pragma unroll
    for (int i = 0; i < 8; ++i) { out[i] = lo[i]; out[i + 8] = hi[i]; }
    return out;
}

__device__ __forceinline__ floatx8 wmma_bf16(bf16x16 a, bf16x16 b, floatx8 c) {
    return __builtin_amdgcn_wmma_f32_16x16x32_bf16(false, a, false, b,
                                                   (short)0, c, false, false);
}

__device__ __forceinline__ floatx8 fzero8() {
    floatx8 z = {0.f, 0.f, 0.f, 0.f, 0.f, 0.f, 0.f, 0.f};
    return z;
}

// ---------------- TDM: DMA a [128 rows x 32 cols] bf16 tile into LDS --------
// D# group1 is built per-kernel (dims uniform); group0 carries lds/global addr.
// pad_enable with pad_interval=16 DWORDs (one 32-elem row) + pad_amount=4
// DWORDs gives an LDS pitch of 40 bf16 (=80B) -> bank-conflict padding in HW.

__device__ __forceinline__ uint32x8 tdm_group1(int K, int nrows) {
    uint32x8 g1;
    g1[0] = (1u << 16)      // data_size = 2 bytes
          | (1u << 20)      // pad_enable
          | (3u << 22)      // pad_interval: 16 DWORDs (one tile row)
          | (3u << 25);     // pad_amount: 4 DWORDs (8 bf16)
    g1[1] = ((unsigned)K & 0xFFFFu) << 16;            // tensor_dim0[15:0]
    g1[2] = (((unsigned)K >> 16) & 0xFFFFu)           // tensor_dim0[31:16]
          | (((unsigned)nrows & 0xFFFFu) << 16);      // tensor_dim1[15:0]
    g1[3] = (((unsigned)nrows >> 16) & 0xFFFFu)       // tensor_dim1[31:16]
          | (32u << 16);                              // tile_dim0 = 32
    g1[4] = 128u;                                     // tile_dim1 = 128
    g1[5] = (unsigned)K;                              // tensor_dim0_stride
    g1[6] = 0u;
    g1[7] = 0u;
    return g1;
}

__device__ __forceinline__ void tdm_load_tile(unsigned int lds_off,
                                              const bf16* gptr, uint32x8 g1) {
    unsigned long long ga = (unsigned long long)(size_t)gptr;
    uint32x4 g0;
    g0[0] = 1u;                                        // count = 1, user mode
    g0[1] = lds_off;                                   // lds_addr (bytes)
    g0[2] = (unsigned int)ga;                          // global_addr[31:0]
    g0[3] = ((unsigned int)(ga >> 32) & 0x01FFFFFFu)   // global_addr[56:32]
          | 0x80000000u;                               // type = 2 ("image")
    asm volatile("tensor_load_to_lds %0, %1" :: "s"(g0), "s"(g1) : "memory");
}

// ---------------- f32 -> bf16 conversion ------------------------------------

__global__ __launch_bounds__(256) void cvt_f32_bf16(const float* __restrict__ in,
                                                    bf16* __restrict__ out, int n) {
    int i = blockIdx.x * 256 + threadIdx.x;
    if (i < n) out[i] = (bf16)in[i];
}

// ---------------- TDM double-buffered WMMA GEMM: C = A * B^T (+bias) --------
// A: [M,K] bf16 row-major.  Bt: [N,K] bf16 row-major (i.e. B transposed).
// Block tile 128x128, BK=32, 8 waves (4x2), wave tile 32x64.
// Wave 0 drives the Tensor Data Mover; TENSORcnt + barriers pipeline the
// next tile's DMA under the current tile's WMMAs.
// mode 0: QKV scatter epilogue (scaled Q, K, V^T head layouts)
// mode 1: f32 output + bias

#define BM 128
#define BN 128
#define BK 32
#define BKP 40   // LDS pitch in elems (80B rows; produced by TDM padding)

__global__ __launch_bounds__(256) void gemm_bf16(
    const bf16* __restrict__ A, const bf16* __restrict__ Bt,
    const float* __restrict__ bias,
    int M, int N, int K, int mode,
    bf16* __restrict__ qbuf, bf16* __restrict__ kbuf, bf16* __restrict__ vtbuf,
    float* __restrict__ outF)
{
    __shared__ bf16 As[2][BM * BKP];
    __shared__ bf16 Bs[2][BN * BKP];

    int tid  = threadIdx.x;
    int lane = tid & 31, wid = tid >> 5;
    int wm = wid >> 1, wn = wid & 1;          // 4 x 2 wave grid
    int r = lane & 15, h = lane >> 4;
    int bm = blockIdx.y * BM, bn = blockIdx.x * BN;

    floatx8 acc[2][4];
#pragma unroll
    for (int fm = 0; fm < 2; ++fm)
#pragma unroll
        for (int fn = 0; fn < 4; ++fn) acc[fm][fn] = fzero8();

    uint32x8 g1a = tdm_group1(K, M);
    uint32x8 g1b = tdm_group1(K, N);
    unsigned int ldsA[2] = { (unsigned int)(size_t)&As[0][0],
                             (unsigned int)(size_t)&As[1][0] };
    unsigned int ldsB[2] = { (unsigned int)(size_t)&Bs[0][0],
                             (unsigned int)(size_t)&Bs[1][0] };
    const bf16* agA = A  + (size_t)bm * K;
    const bf16* agB = Bt + (size_t)bn * K;

    if (wid == 0) {                       // prologue: DMA first tiles
        tdm_load_tile(ldsA[0], agA, g1a);
        tdm_load_tile(ldsB[0], agB, g1b);
    }

    int cur = 0;
    for (int k0 = 0; k0 < K; k0 += BK) {
        int nxt = cur ^ 1;
        bool has_next = (k0 + BK) < K;
        if (wid == 0 && has_next) {       // DMA next tiles under this compute
            tdm_load_tile(ldsA[nxt], agA + k0 + BK, g1a);
            tdm_load_tile(ldsB[nxt], agB + k0 + BK, g1b);
        }
        // in-order TDM: <=2 outstanding means current buffer has landed
        if (has_next)
            __builtin_amdgcn_s_wait_tensorcnt(2);
        else
            __builtin_amdgcn_s_wait_tensorcnt(0);
        __syncthreads();

        bf16x16 af[2], bfr[4];
#pragma unroll
        for (int fm = 0; fm < 2; ++fm)
            af[fm] = frag_a(&As[cur][(wm * 32 + fm * 16) * BKP], BKP);
#pragma unroll
        for (int fn = 0; fn < 4; ++fn)
            bfr[fn] = frag_b(&Bs[cur][(wn * 64 + fn * 16) * BKP], BKP);
#pragma unroll
        for (int fm = 0; fm < 2; ++fm)
#pragma unroll
            for (int fn = 0; fn < 4; ++fn)
                acc[fm][fn] = wmma_bf16(af[fm], bfr[fn], acc[fm][fn]);

        __syncthreads();                  // release cur before it is re-filled
        cur = nxt;
    }

    // Epilogue. C-frag layout: row = i + 8*h (rel), col = r (rel).
    if (mode == 0) {
#pragma unroll
        for (int fn = 0; fn < 4; ++fn) {
            int col = bn + wn * 64 + fn * 16 + r;
            float bv = bias[col];
            int sec = col >> 10;          // 0:q 1:k 2:v
            int p   = col & 1023;
            int hh  = p >> 6;             // head within batch
            int d   = p & 63;
#pragma unroll
            for (int fm = 0; fm < 2; ++fm) {
#pragma unroll
                for (int i = 0; i < 8; ++i) {
                    int row = bm + wm * 32 + fm * 16 + i + 8 * h;  // sb = s*B+b
                    int s = row >> 1, b = row & 1;
                    int n = b * HEADS + hh;
                    float v = acc[fm][fn][i] + bv;
                    if (sec == 0)
                        qbuf[((size_t)n * S_LEN + s) * DHEAD + d] = (bf16)(v * 0.125f);
                    else if (sec == 1)
                        kbuf[((size_t)n * S_LEN + s) * DHEAD + d] = (bf16)v;
                    else
                        vtbuf[((size_t)n * DHEAD + d) * S_LEN + s] = (bf16)v;
                }
            }
        }
    } else {
#pragma unroll
        for (int fn = 0; fn < 4; ++fn) {
            int col = bn + wn * 64 + fn * 16 + r;
            float bv = bias[col];
#pragma unroll
            for (int fm = 0; fm < 2; ++fm) {
#pragma unroll
                for (int i = 0; i < 8; ++i) {
                    int row = bm + wm * 32 + fm * 16 + i + 8 * h;
                    outF[(size_t)row * N + col] = acc[fm][fn][i] + bv;
                }
            }
        }
    }
}

// ---------------- flash attention: one wave = one (head, 16-query tile) -----
// q:  [32][2048][64] bf16 (pre-scaled by 1/sqrt(64))
// k:  [32][2048][64] bf16
// vt: [32][64][2048] bf16 (V transposed)
// ao: [4096][1024]  bf16 (rows = s*B+b, cols = h*64+d)

__global__ __launch_bounds__(256) void attn_flash(
    const bf16* __restrict__ q, const bf16* __restrict__ k,
    const bf16* __restrict__ vt, bf16* __restrict__ ao)
{
    __shared__ bf16 pshare[8][16 * 32];   // per-wave P staging (16x32 bf16)

    int tid = threadIdx.x, lane = tid & 31, wid = tid >> 5;
    int r = lane & 15, h = lane >> 4;
    int task = blockIdx.x * 8 + wid;      // 4096 tasks total
    int head = task >> 7;                 // 0..31
    int qt   = task & 127;
    int m0   = qt * 16;

    const bf16* qh = q  + ((size_t)head * S_LEN + m0) * DHEAD;
    const bf16* kh = k  +  (size_t)head * S_LEN * DHEAD;
    const bf16* vh = vt +  (size_t)head * DHEAD * S_LEN;
    bf16* pb = &pshare[wid][0];

    bf16x16 qa0 = frag_a(qh, DHEAD);        // K-dims 0..31
    bf16x16 qa1 = frag_a(qh + 32, DHEAD);   // K-dims 32..63

    float mi[8], li[8];
    floatx8 o[4];
#pragma unroll
    for (int i = 0; i < 8; ++i) { mi[i] = -1e30f; li[i] = 0.f; }
#pragma unroll
    for (int f = 0; f < 4; ++f) o[f] = fzero8();

    for (int t0 = 0; t0 < S_LEN; t0 += 32) {
        if (t0 + 32 < S_LEN) {            // gfx1250 global_prefetch_b8
            __builtin_prefetch(kh + (size_t)(t0 + 32) * DHEAD, 0, 3);
            __builtin_prefetch(vh + (size_t)t0 + 32, 0, 3);
        }
        // S tiles: keys [t0..t0+15] and [t0+16..t0+31]
        floatx8 s0 = fzero8(), s1 = fzero8();
        s0 = wmma_bf16(qa0, frag_b(kh + (size_t)t0 * DHEAD,            DHEAD), s0);
        s0 = wmma_bf16(qa1, frag_b(kh + (size_t)t0 * DHEAD + 32,       DHEAD), s0);
        s1 = wmma_bf16(qa0, frag_b(kh + (size_t)(t0 + 16) * DHEAD,     DHEAD), s1);
        s1 = wmma_bf16(qa1, frag_b(kh + (size_t)(t0 + 16) * DHEAD + 32,DHEAD), s1);

        // online softmax per output row (row = i + 8*h, stats live in the
        // owning 16-lane half, reduced with xor-shuffles over lanes 1..8)
        float sc[8];
#pragma unroll
        for (int i = 0; i < 8; ++i) {
            float rm = fmaxf(s0[i], s1[i]);
#pragma unroll
            for (int off = 1; off < 16; off <<= 1)
                rm = fmaxf(rm, __shfl_xor(rm, off, 32));
            float mn = fmaxf(mi[i], rm);
            float p0 = __expf(s0[i] - mn);
            float p1 = __expf(s1[i] - mn);
            float rs = p0 + p1;
#pragma unroll
            for (int off = 1; off < 16; off <<= 1)
                rs += __shfl_xor(rs, off, 32);
            sc[i] = __expf(mi[i] - mn);
            li[i] = li[i] * sc[i] + rs;
            mi[i] = mn;
            // stage P tile (16x32, row-major pitch 32) for A-frag reload
            pb[(i + 8 * h) * 32 + r]      = (bf16)p0;
            pb[(i + 8 * h) * 32 + 16 + r] = (bf16)p1;
        }
#pragma unroll
        for (int f = 0; f < 4; ++f)
#pragma unroll
            for (int i = 0; i < 8; ++i) o[f][i] *= sc[i];

        asm volatile("s_wait_dscnt 0" ::: "memory");  // P stores -> P loads
        bf16x16 pa = frag_a(pb, 32);

        // O += P(16x32) * V(32x64): B^T rows are V^T rows (dims), cols = keys
#pragma unroll
        for (int f = 0; f < 4; ++f)
            o[f] = wmma_bf16(pa, frag_b(vh + (size_t)(f * 16) * S_LEN + t0, S_LEN), o[f]);
    }

    int b = head >> 4, hh = head & 15;
#pragma unroll
    for (int i = 0; i < 8; ++i) {
        float inv = 1.0f / li[i];
        int srow = m0 + i + 8 * h;
        size_t rowoff = (size_t)(srow * BATCH + b) * (HEADS * DHEAD);
#pragma unroll
        for (int f = 0; f < 4; ++f)
            ao[rowoff + hh * DHEAD + f * 16 + r] = (bf16)(o[f][i] * inv);
    }
}

// ---------------- host-side launcher -----------------------------------------

extern "C" void kernel_launch(void* const* d_in, const int* in_sizes, int n_in,
                              void* d_out, int out_size, void* d_ws, size_t ws_size,
                              hipStream_t stream) {
    const float* src   = (const float*)d_in[0];   // [2048,2,1024]
    const float* w_in  = (const float*)d_in[1];   // [3072,1024]
    const float* b_in  = (const float*)d_in[2];   // [3072]
    const float* w_out = (const float*)d_in[3];   // [1024,1024]
    const float* b_out = (const float*)d_in[4];   // [1024]
    float* out = (float*)d_out;                   // [2048,2,1024]

    // workspace layout (bytes); attn-out aliases src_bf16 (dead after GEMM1).
    char* ws = (char*)d_ws;
    bf16* srcb  = (bf16*)(ws);                       // 8 MB  [4096,1024]
    bf16* aob   = (bf16*)(ws);                       // alias  [4096,1024]
    bf16* winb  = (bf16*)(ws + ( 8u << 20));         // 6 MB  [3072,1024]
    bf16* woutb = (bf16*)(ws + (14u << 20));         // 2 MB  [1024,1024]
    bf16* qb    = (bf16*)(ws + (16u << 20));         // 8 MB  [32,2048,64]
    bf16* kb    = (bf16*)(ws + (24u << 20));         // 8 MB  [32,2048,64]
    bf16* vtb   = (bf16*)(ws + (32u << 20));         // 8 MB  [32,64,2048]
    // total 40 MB

    cvt_f32_bf16<<<(ROWS * FDIM) / 256, 256, 0, stream>>>(src,   srcb,  ROWS * FDIM);
    cvt_f32_bf16<<<(P3   * FDIM) / 256, 256, 0, stream>>>(w_in,  winb,  P3 * FDIM);
    cvt_f32_bf16<<<(FDIM * FDIM) / 256, 256, 0, stream>>>(w_out, woutb, FDIM * FDIM);

    dim3 g1(P3 / BN, ROWS / BM);       // 24 x 32
    gemm_bf16<<<g1, 256, 0, stream>>>(srcb, winb, b_in, ROWS, P3, FDIM, 0,
                                      qb, kb, vtb, nullptr);

    attn_flash<<<(NH * (S_LEN / 16)) / 8, 256, 0, stream>>>(qb, kb, vtb, aob);

    dim3 g2(FDIM / BN, ROWS / BM);     // 8 x 32
    gemm_bf16<<<g2, 256, 0, stream>>>(aob, woutb, b_out, ROWS, FDIM, FDIM, 1,
                                      nullptr, nullptr, nullptr, out);
}
